// MambaBlock_66786741453369
// MI455X (gfx1250) — matrix operand order repaired and checked by reading, not compiled, not run on hardware
//
#include <hip/hip_runtime.h>

// ---------------- problem constants (from reference) ----------------
#define B_      2
#define L_      2048
#define DMODEL  1024
#define DSTATE  64
#define DCONV   4
#define DINNER  2048
#define DTRANK  64
#define MTOT    (B_ * L_)             // 4096 tokens
#define XDBL_N  (DTRANK + 2 * DSTATE) // 192
#define NCHUNK  16                    // chunks for parallel scan
#define CLEN    (L_ / NCHUNK)         // 128 steps per chunk

typedef _Float16 v16h __attribute__((ext_vector_type(16)));
typedef _Float16 v8h  __attribute__((ext_vector_type(8)));
typedef float    v8f  __attribute__((ext_vector_type(8)));

// ---------------- elementwise helpers ----------------
__global__ void k_f32_to_f16(const float* __restrict__ src,
                             _Float16* __restrict__ dst, int n) {
    int i = blockIdx.x * blockDim.x + threadIdx.x;
    if (i < n) dst[i] = (_Float16)src[i];
}

// src: [K,N] f32 row-major  ->  dst: [N,K] f16 row-major (i.e. B^T for GEMM)
__global__ void k_transpose_f16(const float* __restrict__ src,
                                _Float16* __restrict__ dst, int K, int N) {
    int i = blockIdx.x * blockDim.x + threadIdx.x;
    if (i >= K * N) return;
    int k = i / N, n = i % N;
    dst[(size_t)n * K + k] = (_Float16)src[i];
}

// first DTRANK columns of x_dbl -> f16 [MTOT, DTRANK]
__global__ void k_slice_dtr(const float* __restrict__ xdbl,
                            _Float16* __restrict__ dtr) {
    int i = blockIdx.x * blockDim.x + threadIdx.x;
    if (i >= MTOT * DTRANK) return;
    int m = i / DTRANK, j = i % DTRANK;
    dtr[i] = (_Float16)xdbl[(size_t)m * XDBL_N + j];
}

// causal depthwise conv (k=4) + bias + SiLU.  x is cols [0,DINNER) of xz.
__global__ void k_conv_silu(const float* __restrict__ xz,
                            const float* __restrict__ w,      // [DINNER,1,4]
                            const float* __restrict__ bconv,  // [DINNER]
                            float* __restrict__ xs) {         // [MTOT,DINNER]
    int i = blockIdx.x * blockDim.x + threadIdx.x;
    if (i >= MTOT * DINNER) return;
    int d = i % DINNER;
    int m = i / DINNER;
    int l = m % L_;
    float acc = bconv[d];
#pragma unroll
    for (int k = 0; k < DCONV; ++k) {
        int ls = l - (DCONV - 1) + k;
        if (ls >= 0)
            acc += xz[(size_t)(m - (DCONV - 1) + k) * (2 * DINNER) + d] *
                   w[d * DCONV + k];
    }
    xs[i] = acc / (1.0f + __expf(-acc));   // SiLU
}

// ---------------- WMMA GEMM: C[M,N] = A[M,K] * BT[N,K]^T ----------------
// One wave -> 64x64 output block (4x4 of 16x16 tiles).  launch_bounds(,1)
// lifts the VGPR cap so the 16 f32 accumulators (128 VGPRs) stay resident:
// live set = acc(128) + 4 B frags(32) + 1 A frag(8) + addresses  < 256 VGPRs.
// MODE 0: plain f32 store.  MODE 1: +bias then softplus (for dt).
template <int MODE>
__global__ __launch_bounds__(128, 1) void
k_wmma_gemm(const _Float16* __restrict__ A,
            const _Float16* __restrict__ BT,
            const float* __restrict__ bias,
            float* __restrict__ C,
            int M, int N, int K) {
    const int lane = threadIdx.x & 31;
    const int wave = blockIdx.x * (blockDim.x >> 5) + (threadIdx.x >> 5);
    const int tilesN4 = N >> 6;
    const int tilesTotal = (M >> 6) * tilesN4;
    if (wave >= tilesTotal) return;            // wave-uniform: EXEC stays all-1s
    const int tileM4 = wave / tilesN4;
    const int tileN4 = wave % tilesN4;

    const int r16 = lane & 15;
    const int hi  = lane >> 4;                 // 0 | 1 (lane half)
    const int klo = hi * 8;                    // unused; A loads use kb
    const int kb  = hi * 16;                   // B-fragment K offset

    // single base pointer each; tile offsets are compile-time multiples of
    // 16*K that fold into the 24-bit instruction offset.
    const _Float16* aBase = A  + (size_t)(tileM4 * 64 + r16) * K;
    const _Float16* bBase = BT + (size_t)(tileN4 * 64 + r16) * K + kb;

    v8f acc[4][4] = {};
    for (int k0 = 0; k0 < K; k0 += 32) {
        v16h b[4];
#pragma unroll
        for (int j = 0; j < 4; ++j)
            b[j] = *(const v16h*)(bBase + (size_t)j * 16 * K + k0);
#pragma unroll
        for (int i = 0; i < 4; ++i) {
            v8h alo = *(const v8h*)(aBase + (size_t)i * 16 * K + k0 + kb);
            v8h ahi = *(const v8h*)(aBase + (size_t)i * 16 * K + k0 + kb + 8);
            v16h a;
#pragma unroll
            for (int t = 0; t < 8; ++t) { a[t] = alo[t]; a[t + 8] = ahi[t]; }
#pragma unroll
            for (int j = 0; j < 4; ++j)
                acc[i][j] = __builtin_amdgcn_wmma_f32_16x16x32_f16(
                    false, a, false, b[j], (short)0, acc[i][j], false, false);
        }
    }

#pragma unroll
    for (int i = 0; i < 4; ++i) {
        const int rowBase = tileM4 * 64 + i * 16 + hi * 8;
#pragma unroll
        for (int j = 0; j < 4; ++j) {
            const int col = tileN4 * 64 + j * 16 + r16;
#pragma unroll
            for (int r = 0; r < 8; ++r) {
                float v = acc[i][j][r];
                if (MODE == 1) {                    // bias + softplus
                    v += bias[col];
                    v = (v > 20.0f) ? v : log1pf(__expf(v));
                }
                C[(size_t)(rowBase + r) * N + col] = v;
            }
        }
    }
}

// ---------------- chunked parallel selective scan ----------------
// h[l] = dA[l]*h[l-1] + dB[l] is associative: split L into NCHUNK chunks.
// Pass 1: per chunk, per (d,n): local h_end (from h=0) and the chunk decay
//         prod dA = exp(Ac * sum dt)  (computed from one running sum).
// Combine: serial over 16 chunks per (b,d,n) -> entry state H0 per chunk.
// Pass 2: re-run each chunk from H0, emit fused (+x*D)*silu(z) as f16.
// Parallelism: B*(DINNER/64)*NCHUNK blocks = 2048 waves (vs 128 unchunked).

__global__ __launch_bounds__(64, 1) void
k_scan_part1(const float* __restrict__ dt,
             const float* __restrict__ xs,
             const float* __restrict__ xdbl,
             const float* __restrict__ A_log,
             float* __restrict__ Aprod,   // [B,NCHUNK,DINNER,DSTATE]
             float* __restrict__ Hend) {  // [B,NCHUNK,DINNER,DSTATE]
    __shared__ float sB[DSTATE];
    const int tid = threadIdx.x;                      // 0..63
    const int c   = blockIdx.x % NCHUNK;
    const int dg  = (blockIdx.x / NCHUNK) % (DINNER / 64);
    const int b   = blockIdx.x / (NCHUNK * (DINNER / 64));
    const int d   = dg * 64 + tid;

    float Ac[DSTATE], h[DSTATE];
#pragma unroll
    for (int n = 0; n < DSTATE; ++n) {
        Ac[n] = -__expf(A_log[(size_t)d * DSTATE + n]);
        h[n] = 0.0f;
    }
    float sdt = 0.0f;                                 // running sum of dt

    for (int l = c * CLEN; l < (c + 1) * CLEN; ++l) {
        const size_t m = (size_t)b * L_ + l;
        sB[tid] = xdbl[m * XDBL_N + DTRANK + tid];
        __syncthreads();
        const float dtv = dt[m * DINNER + d];
        const float xv  = xs[m * DINNER + d];
        sdt += dtv;
#pragma unroll
        for (int n = 0; n < DSTATE; ++n) {
            float dA = __expf(Ac[n] * dtv);
            h[n] = dA * h[n] + (dtv * sB[n]) * xv;
        }
        __syncthreads();
    }
    const size_t base = ((size_t)(b * NCHUNK + c) * DINNER + d) * DSTATE;
#pragma unroll
    for (int n = 0; n < DSTATE; ++n) {
        Aprod[base + n] = __expf(Ac[n] * sdt);        // == prod of dA over chunk
        Hend[base + n]  = h[n];
    }
}

__global__ void k_scan_combine(const float* __restrict__ Aprod,
                               const float* __restrict__ Hend,
                               float* __restrict__ H0) {
    int i = blockIdx.x * blockDim.x + threadIdx.x;   // over B*DINNER*DSTATE
    if (i >= B_ * DINNER * DSTATE) return;
    const int b  = i / (DINNER * DSTATE);
    const int dn = i % (DINNER * DSTATE);
    float carry = 0.0f;
    for (int c = 0; c < NCHUNK; ++c) {
        const size_t idx = (size_t)(b * NCHUNK + c) * (DINNER * DSTATE) + dn;
        H0[idx] = carry;
        carry = Aprod[idx] * carry + Hend[idx];
    }
}

__global__ __launch_bounds__(64, 1) void
k_scan_part2(const float* __restrict__ dt,
             const float* __restrict__ xs,
             const float* __restrict__ xdbl,
             const float* __restrict__ xz,     // z in hi half
             const float* __restrict__ A_log,
             const float* __restrict__ Dvec,
             const float* __restrict__ H0,
             _Float16* __restrict__ yg) {      // [MTOT,DINNER]
    __shared__ float sB[DSTATE];
    __shared__ float sC[DSTATE];
    const int tid = threadIdx.x;                      // 0..63
    const int c   = blockIdx.x % NCHUNK;
    const int dg  = (blockIdx.x / NCHUNK) % (DINNER / 64);
    const int b   = blockIdx.x / (NCHUNK * (DINNER / 64));
    const int d   = dg * 64 + tid;

    const size_t base = ((size_t)(b * NCHUNK + c) * DINNER + d) * DSTATE;
    float Ac[DSTATE], h[DSTATE];
#pragma unroll
    for (int n = 0; n < DSTATE; ++n) {
        Ac[n] = -__expf(A_log[(size_t)d * DSTATE + n]);
        h[n]  = H0[base + n];
    }
    const float Dv = Dvec[d];

    for (int l = c * CLEN; l < (c + 1) * CLEN; ++l) {
        const size_t m = (size_t)b * L_ + l;
        sB[tid] = xdbl[m * XDBL_N + DTRANK + tid];
        sC[tid] = xdbl[m * XDBL_N + DTRANK + DSTATE + tid];
        __syncthreads();
        const float dtv = dt[m * DINNER + d];
        const float xv  = xs[m * DINNER + d];
        const float zv  = xz[m * (2 * DINNER) + DINNER + d];
        float y = 0.0f;
#pragma unroll
        for (int n = 0; n < DSTATE; ++n) {
            float dA = __expf(Ac[n] * dtv);
            h[n] = dA * h[n] + (dtv * sB[n]) * xv;
            y += h[n] * sC[n];
        }
        const float g = zv / (1.0f + __expf(-zv));   // silu(z)
        yg[m * DINNER + d] = (_Float16)((y + xv * Dv) * g);
        __syncthreads();
    }
}

// ---------------- host orchestration ----------------
extern "C" void kernel_launch(void* const* d_in, const int* in_sizes, int n_in,
                              void* d_out, int out_size, void* d_ws, size_t ws_size,
                              hipStream_t stream) {
    (void)in_sizes; (void)n_in; (void)out_size; (void)ws_size;
    const float* u     = (const float*)d_in[0];
    const float* W_in  = (const float*)d_in[1];
    const float* convw = (const float*)d_in[2];
    const float* convb = (const float*)d_in[3];
    const float* W_x   = (const float*)d_in[4];
    const float* W_dt  = (const float*)d_in[5];
    const float* b_dt  = (const float*)d_in[6];
    const float* A_log = (const float*)d_in[7];
    const float* Dvec  = (const float*)d_in[8];
    const float* W_out = (const float*)d_in[9];
    float* out = (float*)d_out;

    char* ws = (char*)d_ws;
    size_t off = 0;
    auto alloc = [&](size_t bytes) -> char* {
        off = (off + 255) & ~(size_t)255;
        char* p = ws + off;
        off += bytes;
        return p;
    };

    _Float16* u_h    = (_Float16*)alloc((size_t)MTOT * DMODEL * 2);
    _Float16* WinT   = (_Float16*)alloc((size_t)(2 * DINNER) * DMODEL * 2);
    float*    xz     = (float*)   alloc((size_t)MTOT * 2 * DINNER * 4);
    float*    xs     = (float*)   alloc((size_t)MTOT * DINNER * 4);
    _Float16* xs_h   = (_Float16*)alloc((size_t)MTOT * DINNER * 2);
    _Float16* WxT    = (_Float16*)alloc((size_t)XDBL_N * DINNER * 2);
    float*    xdbl   = (float*)   alloc((size_t)MTOT * XDBL_N * 4);
    _Float16* dtr_h  = (_Float16*)alloc((size_t)MTOT * DTRANK * 2);
    _Float16* WdtT   = (_Float16*)alloc((size_t)DINNER * DTRANK * 2);
    float*    dt     = (float*)   alloc((size_t)MTOT * DINNER * 4);
    _Float16* yg_h   = (_Float16*)alloc((size_t)MTOT * DINNER * 2);
    _Float16* WoutT  = (_Float16*)alloc((size_t)DMODEL * DINNER * 2);
    const size_t scanSz = (size_t)B_ * NCHUNK * DINNER * DSTATE * 4;
    float*    Aprod  = (float*)alloc(scanSz);
    float*    Hend   = (float*)alloc(scanSz);
    float*    H0     = (float*)alloc(scanSz);

    auto gemmBlocks = [](int M, int N) {
        int waves = (M / 64) * (N / 64);        // one wave per 64x64 block
        return (waves + 3) / 4;                 // 4 waves (128 threads) / block
    };
    const int scanGrid = B_ * (DINNER / 64) * NCHUNK;  // 1024 blocks

    int n;
    // 1) stage u and W_in^T as f16
    n = MTOT * DMODEL;
    k_f32_to_f16<<<(n + 255) / 256, 256, 0, stream>>>(u, u_h, n);
    n = DMODEL * 2 * DINNER;
    k_transpose_f16<<<(n + 255) / 256, 256, 0, stream>>>(W_in, WinT, DMODEL, 2 * DINNER);
    // 2) xz = u @ W_in                         [4096 x 4096], K=1024
    k_wmma_gemm<0><<<gemmBlocks(MTOT, 2 * DINNER), 128, 0, stream>>>(
        u_h, WinT, nullptr, xz, MTOT, 2 * DINNER, DMODEL);
    // 3) causal depthwise conv + SiLU -> x_ssm
    n = MTOT * DINNER;
    k_conv_silu<<<(n + 255) / 256, 256, 0, stream>>>(xz, convw, convb, xs);
    // 4) x_dbl = x_ssm @ W_x                   [4096 x 192], K=2048
    k_f32_to_f16<<<(n + 255) / 256, 256, 0, stream>>>(xs, xs_h, n);
    n = DINNER * XDBL_N;
    k_transpose_f16<<<(n + 255) / 256, 256, 0, stream>>>(W_x, WxT, DINNER, XDBL_N);
    k_wmma_gemm<0><<<gemmBlocks(MTOT, XDBL_N), 128, 0, stream>>>(
        xs_h, WxT, nullptr, xdbl, MTOT, XDBL_N, DINNER);
    // 5) dt = softplus(dt_r @ W_dt + b_dt)     [4096 x 2048], K=64
    n = MTOT * DTRANK;
    k_slice_dtr<<<(n + 255) / 256, 256, 0, stream>>>(xdbl, dtr_h);
    n = DTRANK * DINNER;
    k_transpose_f16<<<(n + 255) / 256, 256, 0, stream>>>(W_dt, WdtT, DTRANK, DINNER);
    k_wmma_gemm<1><<<gemmBlocks(MTOT, DINNER), 128, 0, stream>>>(
        dtr_h, WdtT, b_dt, dt, MTOT, DINNER, DTRANK);
    // 6) chunked parallel scan + skip + gate -> f16 operand
    k_scan_part1<<<scanGrid, 64, 0, stream>>>(dt, xs, xdbl, A_log, Aprod, Hend);
    n = B_ * DINNER * DSTATE;
    k_scan_combine<<<(n + 255) / 256, 256, 0, stream>>>(Aprod, Hend, H0);
    k_scan_part2<<<scanGrid, 64, 0, stream>>>(dt, xs, xdbl, xz, A_log, Dvec, H0, yg_h);
    // 7) out = y_gated @ W_out                 [4096 x 1024], K=2048
    n = DINNER * DMODEL;
    k_transpose_f16<<<(n + 255) / 256, 256, 0, stream>>>(W_out, WoutT, DINNER, DMODEL);
    k_wmma_gemm<0><<<gemmBlocks(MTOT, DMODEL), 128, 0, stream>>>(
        yg_h, WoutT, nullptr, out, MTOT, DMODEL, DINNER);
}
